// LSTM_23493471109372
// MI455X (gfx1250) — matrix operand order repaired and checked by reading, not compiled, not run on hardware
//
#include <hip/hip_runtime.h>

// ---------------------------------------------------------------------------
// LSTM on MI455X (gfx1250): bf16 WMMA recurrent GEMM, fused gate update.
// B=512, T=256, I=64, H=512, O=64.  K = I + H = 576.
// ---------------------------------------------------------------------------

typedef __attribute__((ext_vector_type(16))) __bf16 v16bf;
typedef __attribute__((ext_vector_type(8)))  float  v8f;

#define BATCH 512
#define TSTEPS 256
#define IDIM 64
#define HDIM 512
#define ODIM 64
#define KDIM 576            // IDIM + HDIM
#define G4H  2048           // 4*HDIM

// ---- bf16 helpers (self-contained, RNE) -----------------------------------
static __device__ __forceinline__ unsigned short f2bf(float f) {
    union { float f; unsigned u; } v; v.f = f;
    unsigned r = v.u + 0x7FFFu + ((v.u >> 16) & 1u);
    return (unsigned short)(r >> 16);
}
static __device__ __forceinline__ float bf2f(unsigned short b) {
    union { unsigned u; float f; } v; v.u = ((unsigned)b) << 16; return v.f;
}
static __device__ __forceinline__ float sigmoidf_(float x) {
    return 1.0f / (1.0f + __expf(-x));
}

// ---- fragment loaders per CDNA5 ISA VGPR layouts --------------------------
// A (16x32 bf16, M x K): lane m = lane&15, half = lane>>4;
//   vector elems 0..7  = K[k0 + half*8      .. +7]
//   vector elems 8..15 = K[k0 + 16 + half*8 .. +7]
union FragU { uint4 u[2]; v16bf v; };

static __device__ __forceinline__ v16bf load_frag_A(const unsigned short* A, int lda,
                                                    int m0, int k0, int lane) {
    int m    = m0 + (lane & 15);
    int half = lane >> 4;
    const unsigned short* p = A + m * lda + k0 + half * 8;
    FragU r;
    r.u[0] = *(const uint4*)(p);
    r.u[1] = *(const uint4*)(p + 16);
    return r.v;
}

// B (32x16 bf16, K x N) from row-major W[n][k] (B = W^T):
// lane n = lane&15, half = lane>>4; elems 0..7 = K[k0 + half*8 .. +7], elems 8..15 = K[k0 + 16 + half*8 .. +7]
static __device__ __forceinline__ v16bf load_frag_B(const unsigned short* W, int ldw,
                                                    int n0, int k0, int lane) {
    int n    = n0 + (lane & 15);
    int half = lane >> 4;
    const unsigned short* p = W + n * ldw + k0 + half * 8;
    FragU r;
    r.u[0] = *(const uint4*)(p);
    r.u[1] = *(const uint4*)(p + 16);
    return r.v;
}

// ---------------------------------------------------------------------------
// Setup: pack [W_ih | W_hh] -> bf16 Wc[2048][576]; bsum = b_ih + b_hh;
// zero c (f32) and h0 (bf16).  Grid: 4608 x 256 = 1,179,648 threads.
// ---------------------------------------------------------------------------
__global__ void lstm_setup(const float* __restrict__ W_ih,
                           const float* __restrict__ W_hh,
                           const float* __restrict__ b_ih,
                           const float* __restrict__ b_hh,
                           unsigned short* __restrict__ Wc,
                           float* __restrict__ bsum,
                           float* __restrict__ cbuf,
                           unsigned short* __restrict__ h0) {
    int idx = blockIdx.x * blockDim.x + threadIdx.x;   // 0 .. 2048*576-1
    int n = idx / KDIM;
    int k = idx - n * KDIM;
    float w = (k < IDIM) ? W_ih[n * IDIM + k] : W_hh[n * HDIM + (k - IDIM)];
    Wc[idx] = f2bf(w);
    if (idx < G4H)           bsum[idx] = b_ih[idx] + b_hh[idx];
    if (idx < BATCH * HDIM) { cbuf[idx] = 0.0f; h0[idx] = 0; }
}

// ---------------------------------------------------------------------------
// One recurrent step.  Block = 256 threads (8 wave32).
// Workgroup tile: 32 batch rows (blockIdx.y) x 32 hidden cols (blockIdx.x),
// covering gate columns {n, n+512, n+1024, n+1536}.
// Wave w: gate = w>>1, n-subtile = w&1; loops over 2 M-subtiles.
// ---------------------------------------------------------------------------
__global__ void __launch_bounds__(256)
lstm_step(const float* __restrict__ x,          // [B, T, I]
          const unsigned short* __restrict__ Wc, // [2048, 576] bf16
          const float* __restrict__ bsum,        // [2048]
          float* __restrict__ cbuf,              // [B, H]
          const unsigned short* __restrict__ hin,// [B, H] bf16
          unsigned short* __restrict__ hout,     // [B, H] bf16
          int t) {
    __shared__ unsigned short Atile[32 * KDIM];      // 36 KB: [x_t | h] bf16
    __shared__ float gbuf[4][32][32];                // 16 KB: i,f,g,o tiles

    const int tid  = threadIdx.x;
    const int lane = tid & 31;
    const int wave = tid >> 5;
    const int m0   = blockIdx.y * 32;                // batch-row base
    const int n0   = blockIdx.x * 32;                // hidden-col base

    // ---- stage A = [x_t (f32->bf16) | h (bf16 copy)] into LDS -------------
    for (int i = tid; i < 32 * IDIM; i += 256) {     // 2048 elems, 8/thread
        int r = i >> 6, c = i & 63;
        Atile[r * KDIM + c] = f2bf(x[((size_t)(m0 + r) * TSTEPS + t) * IDIM + c]);
    }
    for (int i = tid; i < 32 * (HDIM / 8); i += 256) { // 2048 uint4, 8/thread
        int r = i >> 6, c = i & 63;                    // 64 chunks per row
        ((uint4*)(Atile + r * KDIM + IDIM))[c] =
            ((const uint4*)(hin + (size_t)(m0 + r) * HDIM))[c];
    }
    __syncthreads();

    // ---- WMMA GEMM: gates(32 x 4x16-per-wave) over K = 576 ----------------
    const int gate = wave >> 1;                      // 0..3  (i,f,g,o)
    const int nt   = wave & 1;                       // 0..1
    const int nrow = gate * HDIM + n0 + nt * 16;     // Wc row base for B frag

    v8f acc0 = {};
    v8f acc1 = {};
    for (int k0 = 0; k0 < KDIM; k0 += 32) {
        v16bf b  = load_frag_B(Wc, KDIM, nrow, k0, lane);
        v16bf a0 = load_frag_A(Atile, KDIM, 0, k0, lane);
        v16bf a1 = load_frag_A(Atile, KDIM, 16, k0, lane);
        acc0 = __builtin_amdgcn_wmma_f32_16x16x32_bf16(
            false, a0, false, b, (short)0, acc0, false, false);
        acc1 = __builtin_amdgcn_wmma_f32_16x16x32_bf16(
            false, a1, false, b, (short)0, acc1, false, false);
    }

    // ---- scatter D tiles to LDS (C/D layout: m = vgpr + 8*half, n = lane&15)
    {
        int n = lane & 15, half = lane >> 4;
        #pragma unroll
        for (int r = 0; r < 8; ++r) {
            gbuf[gate][half * 8 + r][nt * 16 + n]      = acc0[r];
            gbuf[gate][16 + half * 8 + r][nt * 16 + n] = acc1[r];
        }
    }
    __syncthreads();

    // ---- fused gate nonlinearity + c/h update (1024 elems, 4/thread) ------
    for (int e = tid; e < 32 * 32; e += 256) {
        int r  = e >> 5;                 // local batch row
        int cc = e & 31;                 // local hidden col
        size_t m  = (size_t)(m0 + r);
        int    hc = n0 + cc;
        float ig = gbuf[0][r][cc] + bsum[0 * HDIM + hc];
        float fg = gbuf[1][r][cc] + bsum[1 * HDIM + hc];
        float gg = gbuf[2][r][cc] + bsum[2 * HDIM + hc];
        float og = gbuf[3][r][cc] + bsum[3 * HDIM + hc];
        float iv = sigmoidf_(ig);
        float fv = sigmoidf_(fg);
        float gv = tanhf(gg);
        float ov = sigmoidf_(og);
        float cprev = cbuf[m * HDIM + hc];
        float cnew  = fv * cprev + iv * gv;
        cbuf[m * HDIM + hc] = cnew;
        hout[m * HDIM + hc] = f2bf(ov * tanhf(cnew));
    }
}

// ---------------------------------------------------------------------------
// Head: out[m][oc] = b_fc[oc] + sum_j h[m][j] * W_fc[oc][j]   (fp32, tiny)
// ---------------------------------------------------------------------------
__global__ void lstm_head(const unsigned short* __restrict__ h,
                          const float* __restrict__ W_fc,
                          const float* __restrict__ b_fc,
                          float* __restrict__ out) {
    int idx = blockIdx.x * blockDim.x + threadIdx.x;  // 32768
    int m = idx >> 6, oc = idx & 63;
    const unsigned short* hr = h + (size_t)m * HDIM;
    const float* wr = W_fc + oc * HDIM;
    float s = b_fc[oc];
    for (int j = 0; j < HDIM; ++j) s += bf2f(hr[j]) * wr[j];
    out[idx] = s;
}

// ---------------------------------------------------------------------------
extern "C" void kernel_launch(void* const* d_in, const int* in_sizes, int n_in,
                              void* d_out, int out_size, void* d_ws, size_t ws_size,
                              hipStream_t stream) {
    const float* x    = (const float*)d_in[0];
    const float* W_ih = (const float*)d_in[1];
    const float* W_hh = (const float*)d_in[2];
    const float* b_ih = (const float*)d_in[3];
    const float* b_hh = (const float*)d_in[4];
    const float* W_fc = (const float*)d_in[5];
    const float* b_fc = (const float*)d_in[6];

    // Workspace layout (~4.46 MB total, all offsets 128B-aligned):
    char* ws = (char*)d_ws;
    unsigned short* Wc   = (unsigned short*)(ws);                 // 2,359,296 B
    float*          bsum = (float*)         (ws + 2359296);       //     8,192 B
    float*          cbuf = (float*)         (ws + 2367488);       // 1,048,576 B
    unsigned short* h0   = (unsigned short*)(ws + 3416064);       //   524,288 B
    unsigned short* h1   = (unsigned short*)(ws + 3940352);       //   524,288 B

    lstm_setup<<<dim3(4608), dim3(256), 0, stream>>>(
        W_ih, W_hh, b_ih, b_hh, Wc, bsum, cbuf, h0);

    // 256 sequential steps; double-buffer h to avoid cross-WG races.
    for (int t = 0; t < TSTEPS; ++t) {
        unsigned short* hin  = (t & 1) ? h1 : h0;
        unsigned short* hout = (t & 1) ? h0 : h1;
        lstm_step<<<dim3(HDIM / 32, BATCH / 32), dim3(256), 0, stream>>>(
            x, Wc, bsum, cbuf, hin, hout, t);
    }
    // T even -> final h lives in h0.
    lstm_head<<<dim3((BATCH * ODIM) / 256), dim3(256), 0, stream>>>(
        h0, W_fc, b_fc, (float*)d_out);
}
